// LMUCell_33586644255223
// MI455X (gfx1250) — matrix operand order repaired and checked by reading, not compiled, not run on hardware
//
#include <hip/hip_runtime.h>
#include <hip/hip_bf16.h>

#define MEMORY_D 32
#define ORDER    64
#define HIDDEN   512
#define BATCH    64
#define TLEN     1024
#define D_IN     128
#define KDIM     (MEMORY_D*ORDER)   /* 2048 */
#define TC       64                 /* time chunk for m-scan */
#define NCHUNK   (TLEN/TC)          /* 16 */

typedef __attribute__((ext_vector_type(16))) __bf16 bf16x16;
typedef __attribute__((ext_vector_type(8)))  __bf16 bf16x8;
typedef __attribute__((ext_vector_type(8)))  float  f32x8;

__device__ __forceinline__ unsigned short f2bf(float x) {
    union { float f; unsigned u; } v; v.f = x;
    unsigned r = v.u + 0x7FFFu + ((v.u >> 16) & 1u);   // RNE
    return (unsigned short)(r >> 16);
}

// ---------------------------------------------------------------- u = x @ kernel
__global__ __launch_bounds__(256) void u_kernel(const float* __restrict__ x,
                                                const float* __restrict__ kern,
                                                float* __restrict__ u)
{
    int tid = blockIdx.x * 256 + threadIdx.x;  // layout u[t][b][c]
    int c = tid & 31;
    int b = (tid >> 5) & 63;
    int t = tid >> 11;
    const float* xr = x + ((size_t)b * TLEN + t) * D_IN;
    float acc = 0.f;
#pragma unroll 8
    for (int d = 0; d < D_IN; ++d)
        acc = __builtin_fmaf(xr[d], kern[d * MEMORY_D + c], acc);
    u[tid] = acc;
}

// ------------------------------------------- f32 [K][N] -> bf16 [N][K] transpose
__global__ __launch_bounds__(256) void transpose_bf16(const float* __restrict__ in,
                                                      unsigned short* __restrict__ out,
                                                      int K, int N)
{
    int tid = blockIdx.x * 256 + threadIdx.x;
    if (tid >= K * N) return;
    int n = tid / K, k = tid - n * K;
    out[tid] = f2bf(in[k * N + n]);
}

// ------------------- m-scan over one time chunk: 2048 independent rows, fp32 state
__global__ __launch_bounds__(256) void mscan_kernel(const float* __restrict__ u,
                                                    const float* __restrict__ A,
                                                    const float* __restrict__ Bv,
                                                    float* __restrict__ mstate,
                                                    unsigned short* __restrict__ mchunk,
                                                    int t0, int first)
{
    int r = blockIdx.x * 256 + threadIdx.x;   // r = b*32 + c, 2048 rows total
    int b = r >> 5, c = r & 31;
    float s[ORDER];
    if (first) {
#pragma unroll
        for (int p = 0; p < ORDER; ++p) s[p] = 0.f;
    } else {
#pragma unroll
        for (int p = 0; p < ORDER; ++p) s[p] = mstate[r * ORDER + p];
    }
#pragma unroll 1
    for (int tl = 0; tl < TC; ++tl) {
        float uv = u[((size_t)(t0 + tl) * BATCH + b) * MEMORY_D + c];
        float ns[ORDER];
#pragma unroll
        for (int p = 0; p < ORDER; ++p) ns[p] = uv * Bv[p];
#pragma unroll
        for (int o = 0; o < ORDER; ++o) {
            float so = s[o];
#pragma unroll
            for (int p = 0; p < ORDER; ++p)
                ns[p] = __builtin_fmaf(so, A[o * ORDER + p], ns[p]);  // A uniform -> s_loads
        }
#pragma unroll
        for (int p = 0; p < ORDER; ++p) s[p] = ns[p];
        unsigned* dst = (unsigned*)(mchunk + ((size_t)tl * BATCH + b) * KDIM + c * ORDER);
#pragma unroll
        for (int p = 0; p < ORDER; p += 2)
            dst[p >> 1] = (unsigned)f2bf(s[p]) | ((unsigned)f2bf(s[p + 1]) << 16);
    }
#pragma unroll
    for (int p = 0; p < ORDER; ++p) mstate[r * ORDER + p] = s[p];
}

// --------------- fragment loaders (contiguous-K 16-bit A 16x32 layout; B from [N][K] bf16)
__device__ __forceinline__ bf16x16 load_a_frag(const unsigned short* base, int ld,
                                               int row, int k0, int lm, int lh)
{
    const unsigned short* p = base + (size_t)(row + lm) * ld + k0 + lh * 16;
    bf16x8 lo = *(const bf16x8*)p;          // K = k0 + lh*16 .. +7
    bf16x8 hi = *(const bf16x8*)(p + 8);    // K = k0 + lh*16 + 8 .. +15
    bf16x16 a;
#pragma unroll
    for (int i = 0; i < 8; ++i) { a[i] = lo[i]; a[i + 8] = hi[i]; }
    return a;
}
__device__ __forceinline__ bf16x16 load_b_frag(const unsigned short* base, int ld,
                                               int col, int k0, int lm, int lh)
{
    // lane holds 16 consecutive K for column N=lm; lanes 16-31 take K+16..31
    return *(const bf16x16*)(base + (size_t)(col + lm) * ld + k0 + lh * 16);
}

// --------------------- P[t][b][:] = m_flat @ Wh + bh   (bf16 WMMA, 64x64 per wave)
__global__ __launch_bounds__(256) void gemm_p_kernel(const unsigned short* __restrict__ mchunk,
                                                     const unsigned short* __restrict__ WhT,
                                                     const float* __restrict__ bh,
                                                     float* __restrict__ out, int t0)
{
    int wave = threadIdx.x >> 5, lane = threadIdx.x & 31;
    int lm = lane & 15, lh = lane >> 4;
    int mBase = blockIdx.x * 64;           // rows in chunk (TC*BATCH = 4096)
    int nBase = wave * 64;                 // 8 waves cover N = 512
    f32x8 c[4][4] = {};
#pragma unroll 1
    for (int k0 = 0; k0 < KDIM; k0 += 32) {
        bf16x16 a[4], bb[4];
#pragma unroll
        for (int mi = 0; mi < 4; ++mi)
            a[mi] = load_a_frag(mchunk, KDIM, mBase + 16 * mi, k0, lm, lh);
#pragma unroll
        for (int ni = 0; ni < 4; ++ni)
            bb[ni] = load_b_frag(WhT, KDIM, nBase + 16 * ni, k0, lm, lh);
#pragma unroll
        for (int mi = 0; mi < 4; ++mi)
#pragma unroll
            for (int ni = 0; ni < 4; ++ni)
                c[mi][ni] = __builtin_amdgcn_wmma_f32_16x16x32_bf16(
                    false, a[mi], false, bb[ni], (short)0, c[mi][ni], false, false);
    }
#pragma unroll
    for (int mi = 0; mi < 4; ++mi)
#pragma unroll
        for (int ni = 0; ni < 4; ++ni)
#pragma unroll
            for (int rr = 0; rr < 8; ++rr) {
                int m = mBase + 16 * mi + rr + 8 * lh;   // chunk row -> (tl, b)
                int n = nBase + 16 * ni + lm;
                int tl = m >> 6, b = m & 63;
                out[((size_t)b * TLEN + (t0 + tl)) * HIDDEN + n] = c[mi][ni][rr] + bh[n];
            }
}

// ------- sequential h-recurrence: one persistent WG, h resident in LDS as bf16
__global__ __launch_bounds__(512) void hscan_kernel(const unsigned short* __restrict__ UhT,
                                                    float* __restrict__ out)
{
    __shared__ alignas(16) unsigned short hbuf[BATCH * HIDDEN];   // 64 KB
    int tid = threadIdx.x;
    int wave = tid >> 5, lane = tid & 31;
    int lm = lane & 15, lh = lane >> 4;
    for (int i = tid; i < BATCH * HIDDEN / 2; i += 512) ((unsigned*)hbuf)[i] = 0u;
    __syncthreads();
    int nBase = wave * 32;                 // 16 waves x 32 cols = HIDDEN
#pragma unroll 1
    for (int t = 0; t < TLEN; ++t) {
        f32x8 c[4][2] = {};
#pragma unroll 1
        for (int k0 = 0; k0 < HIDDEN; k0 += 32) {
            bf16x16 a[4], bb[2];
#pragma unroll
            for (int mi = 0; mi < 4; ++mi) {
                int idx = (16 * mi + lm) * HIDDEN + k0 + lh * 16;
                bf16x8 lo = *(const bf16x8*)&hbuf[idx];
                bf16x8 hi = *(const bf16x8*)&hbuf[idx + 8];
#pragma unroll
                for (int i = 0; i < 8; ++i) { a[mi][i] = lo[i]; a[mi][i + 8] = hi[i]; }
            }
#pragma unroll
            for (int ni = 0; ni < 2; ++ni)
                bb[ni] = *(const bf16x16*)(UhT + (size_t)(nBase + 16 * ni + lm) * HIDDEN
                                               + k0 + lh * 16);
#pragma unroll
            for (int mi = 0; mi < 4; ++mi)
#pragma unroll
                for (int ni = 0; ni < 2; ++ni)
                    c[mi][ni] = __builtin_amdgcn_wmma_f32_16x16x32_bf16(
                        false, a[mi], false, bb[ni], (short)0, c[mi][ni], false, false);
        }
        __syncthreads();   // all reads of h_{t-1} done before overwrite
#pragma unroll
        for (int mi = 0; mi < 4; ++mi)
#pragma unroll
            for (int ni = 0; ni < 2; ++ni)
#pragma unroll
                for (int rr = 0; rr < 8; ++rr) {
                    int b = 16 * mi + rr + 8 * lh;           // M dim == batch
                    int n = nBase + 16 * ni + lm;
                    size_t gi = ((size_t)b * TLEN + t) * HIDDEN + n;
                    float h = tanhf(c[mi][ni][rr] + out[gi]); // out holds m.Wh + bh
                    out[gi] = h;
                    hbuf[b * HIDDEN + n] = f2bf(h);
                }
        __syncthreads();   // h_t visible before step t+1 reads
    }
}

extern "C" void kernel_launch(void* const* d_in, const int* in_sizes, int n_in,
                              void* d_out, int out_size, void* d_ws, size_t ws_size,
                              hipStream_t stream)
{
    const float* x    = (const float*)d_in[0];
    const float* kern = (const float*)d_in[1];
    const float* A    = (const float*)d_in[2];
    const float* Bv   = (const float*)d_in[3];
    const float* Wh   = (const float*)d_in[4];
    const float* Uh   = (const float*)d_in[5];
    const float* bh   = (const float*)d_in[6];
    float* out = (float*)d_out;

    // workspace carve (all 256B aligned): u 8MiB | mstate 512KiB | WhT 2MiB | UhT 512KiB | mchunk 16MiB
    char* ws = (char*)d_ws;
    float*          u      = (float*)(ws);
    float*          mstate = (float*)(ws + (8u << 20));
    unsigned short* WhT    = (unsigned short*)(ws + (8u << 20) + (512u << 10));
    unsigned short* UhT    = (unsigned short*)(ws + (8u << 20) + (512u << 10) + (2u << 20));
    unsigned short* mchunk = (unsigned short*)(ws + (8u << 20) + (512u << 10) + (2u << 20) + (512u << 10));

    u_kernel<<<(TLEN * BATCH * MEMORY_D) / 256, 256, 0, stream>>>(x, kern, u);
    transpose_bf16<<<(KDIM * HIDDEN + 255) / 256, 256, 0, stream>>>(Wh, WhT, KDIM, HIDDEN);
    transpose_bf16<<<(HIDDEN * HIDDEN + 255) / 256, 256, 0, stream>>>(Uh, UhT, HIDDEN, HIDDEN);

    for (int ci = 0; ci < NCHUNK; ++ci) {
        mscan_kernel<<<(BATCH * MEMORY_D) / 256, 256, 0, stream>>>(
            u, A, Bv, mstate, mchunk, ci * TC, ci == 0 ? 1 : 0);
        gemm_p_kernel<<<(TC * BATCH) / 64, 256, 0, stream>>>(mchunk, WhT, bh, out, ci * TC);
    }
    hscan_kernel<<<1, 512, 0, stream>>>(UhT, out);
}
